// Ernie4_5_VLMoeDecoderLayer_9294309228909
// MI455X (gfx1250) — matrix operand order repaired and checked by reading, not compiled, not run on hardware
//
#include <hip/hip_runtime.h>
#include <hip/hip_bf16.h>
#include <stdint.h>

#define S_    2048
#define H_    1024
#define NH_   8
#define NKV_  2
#define D_    128
#define E_    64
#define TOPK_ 6
#define IT_   512
#define IV_   256
#define IS_   1024
#define CAP_  (S_*TOPK_)
#define THETA_ 500000.0f

typedef __bf16 bf16;
typedef __attribute__((ext_vector_type(16))) __bf16 v16bf;
typedef __attribute__((ext_vector_type(8)))  __bf16 v8bf;
typedef __attribute__((ext_vector_type(8)))  float  v8f;

// ---------------------------------------------------------------- WMMA core
__device__ __forceinline__ v8f wmma_bf16(v16bf a, v16bf b, v8f c) {
  return __builtin_amdgcn_wmma_f32_16x16x32_bf16(false, a, false, b, (short)0, c, false, false);
}

// A fragment (16x32 bf16): lanes 0-15 row M=lane, elems 0..7 = K[kh..kh+7],
// elems 8..15 = K[16+kh..], kh = (lane>>4)*8.
__device__ __forceinline__ v16bf load_a_frag(const bf16* sA, int stride, int mbase, int lane) {
  const bf16* p = sA + (size_t)(mbase + (lane & 15)) * stride;
  int kh = (lane >> 4) * 8;
  v8bf lo = *(const v8bf*)(p + kh);
  v8bf hi = *(const v8bf*)(p + 16 + kh);
  v16bf a;
#pragma unroll
  for (int i = 0; i < 8; ++i) { a[i] = lo[i]; a[i + 8] = hi[i]; }
  return a;
}

// B fragment (32x16 bf16): N = lane&15, elems 0..7 = K[kh..kh+7],
// elems 8..15 = K[16+kh..], kh = (lane>>4)*8.
__device__ __forceinline__ v16bf load_b_frag(const bf16* sB, int stride, int nbase, int lane) {
  const bf16* p = sB + nbase + (lane & 15);
  int kh = (lane >> 4) * 8;
  v16bf b;
#pragma unroll
  for (int i = 0; i < 8; ++i) {
    b[i] = p[(size_t)(kh + i) * stride];
    b[i + 8] = p[(size_t)(16 + kh + i) * stride];
  }
  return b;
}

__device__ __forceinline__ float siluf(float x) {
  // x * rcp(1+exp(-x)): one v_exp + one v_rcp instead of an IEEE divide macro
  return x * __builtin_amdgcn_rcpf(1.f + __expf(-x));
}

// ------------------------------------------------ CDNA5 async global->LDS DMA
// GLOBAL_LOAD_ASYNC_TO_LDS_B128: per-lane 16B copy, tracked by ASYNCcnt.
__device__ __forceinline__ void async_cp16(void* lds_ptr, const void* gptr) {
  uint32_t ldsa = (uint32_t)(uintptr_t)lds_ptr;  // LDS aperture: addr[31:0] = LDS offset
  asm volatile("global_load_async_to_lds_b128 %0, %1, off"
               :: "v"(ldsa), "v"(gptr) : "memory");
}
__device__ __forceinline__ void async_wait0() {
  asm volatile("s_wait_asynccnt 0x0" ::: "memory");
}

// ---------------------------------------------------------------- generic GEMM
// C[M,N] = A_bf16[M,K] * B[K,N]; block tile 128x64, 8 waves (4x2), wave 32x32.
template<bool B_F32, bool C_BF16, bool CAUSAL>
__launch_bounds__(256)
__global__ void k_gemm(const bf16* __restrict__ A, int lda,
                       const void* __restrict__ Bv, int ldb,
                       void* __restrict__ Cv, int ldc,
                       int M, int N, int K)
{
  const int m0 = blockIdx.y * 128;
  const int n0 = blockIdx.x * 64;
  if (CAUSAL && n0 > m0 + 127) return;   // entire tile above causal diagonal
  __shared__ __align__(32) bf16 sA[128 * 40];
  __shared__ __align__(32) bf16 sB[32 * 80];
  const int tid = threadIdx.x, lane = tid & 31, wid = tid >> 5;
  const int mw = (wid >> 1) * 32, nw = (wid & 1) * 32;
  v8f acc[2][2] = {};
  for (int k0 = 0; k0 < K; k0 += 32) {
    { // A tile 128x32 bf16: async DMA straight into LDS
      const int r = tid >> 1, kc = (tid & 1) * 16;
      const bf16* g = A + (size_t)(m0 + r) * lda + (k0 + kc);
      async_cp16(&sA[r * 40 + kc],     g);
      async_cp16(&sA[r * 40 + kc + 8], g + 8);
    }
    { // B tile 32x64 (convert f32->bf16 on the fly if needed)
      const int r = tid >> 3, c = (tid & 7) * 8;
      if (B_F32) {
        const float* g = (const float*)Bv + (size_t)(k0 + r) * ldb + (n0 + c);
        float4 f0 = *(const float4*)g, f1 = *(const float4*)(g + 4);
        bf16* d = &sB[r * 80 + c];
        d[0] = (bf16)f0.x; d[1] = (bf16)f0.y; d[2] = (bf16)f0.z; d[3] = (bf16)f0.w;
        d[4] = (bf16)f1.x; d[5] = (bf16)f1.y; d[6] = (bf16)f1.z; d[7] = (bf16)f1.w;
        if (k0 + 32 < K)  // keep GL2 ahead of the weight stream
          __builtin_prefetch((const float*)Bv + (size_t)(k0 + 32 + r) * ldb + (n0 + c), 0, 1);
      } else {
        const bf16* g = (const bf16*)Bv + (size_t)(k0 + r) * ldb + (n0 + c);
        async_cp16(&sB[r * 80 + c], g);
      }
    }
    async_wait0();
    __syncthreads();
    v16bf a0 = load_a_frag(sA, 40, mw,      lane);
    v16bf a1 = load_a_frag(sA, 40, mw + 16, lane);
    v16bf b0 = load_b_frag(sB, 80, nw,      lane);
    v16bf b1 = load_b_frag(sB, 80, nw + 16, lane);
    acc[0][0] = wmma_bf16(a0, b0, acc[0][0]);
    acc[0][1] = wmma_bf16(a0, b1, acc[0][1]);
    acc[1][0] = wmma_bf16(a1, b0, acc[1][0]);
    acc[1][1] = wmma_bf16(a1, b1, acc[1][1]);
    __syncthreads();
  }
  const int rb = (lane >> 4) * 8, cl = lane & 15;
#pragma unroll
  for (int mi = 0; mi < 2; ++mi)
#pragma unroll
    for (int ni = 0; ni < 2; ++ni) {
      const int gr = m0 + mw + mi * 16 + rb;
      const int gc = n0 + nw + ni * 16 + cl;
#pragma unroll
      for (int r = 0; r < 8; ++r) {
        float v = acc[mi][ni][r];
        if (C_BF16) ((bf16*)Cv)[(size_t)(gr + r) * ldc + gc] = (bf16)v;
        else        ((float*)Cv)[(size_t)(gr + r) * ldc + gc] = v;
      }
    }
}

// ---------------------------------------------------------------- fused gate/up (+SiLU*) GEMM
// Rows gathered via token list (or identity if idx==null). Block 64 rows x 64 cols,
// 8 waves (2x4), wave 32x16; two accumulators per B matrix.
__launch_bounds__(256)
__global__ void k_gateup(const bf16* __restrict__ X,
                         const int* __restrict__ idx, const int* __restrict__ offs,
                         const int* __restrict__ counts, int fixedCount,
                         const float* __restrict__ Wg, const float* __restrict__ Wu,
                         long long wstride, bf16* __restrict__ act, int NI, int K)
{
  const int e = blockIdx.z;
  const int pairBase = offs ? offs[e] : 0;
  const int count = counts ? counts[e] : fixedCount;
  const int row0 = blockIdx.y * 64;
  if (row0 >= count) return;
  const int n0 = blockIdx.x * 64;
  const float* Bg = Wg + (size_t)e * wstride;
  const float* Bu = Wu + (size_t)e * wstride;
  __shared__ __align__(32) bf16 sA[64 * 40];
  __shared__ __align__(32) bf16 sBg[32 * 80];
  __shared__ __align__(32) bf16 sBu[32 * 80];
  __shared__ int sTok[64];
  const int tid = threadIdx.x, lane = tid & 31, wid = tid >> 5;
  if (tid < 64) {
    int r = row0 + tid;
    int rc = (r < count) ? r : (count - 1);
    sTok[tid] = idx ? idx[pairBase + rc] : rc;
  }
  __syncthreads();
  const int mw = (wid >> 2) * 32, nw = (wid & 3) * 16;
  v8f ag[2] = {}, au[2] = {};
  for (int k0 = 0; k0 < K; k0 += 32) {
    { // gathered A tile 64x32: async DMA into LDS
      const int r = tid >> 2, kc = (tid & 3) * 8;
      const bf16* g = X + (size_t)sTok[r] * K + (k0 + kc);
      async_cp16(&sA[r * 40 + kc], g);
    }
    { // two B tiles 32x64 f32->bf16
      const int r = tid >> 3, c = (tid & 7) * 8;
      const float* gg = Bg + (size_t)(k0 + r) * NI + (n0 + c);
      const float* gu = Bu + (size_t)(k0 + r) * NI + (n0 + c);
      float4 a0 = *(const float4*)gg, a1 = *(const float4*)(gg + 4);
      float4 b0 = *(const float4*)gu, b1 = *(const float4*)(gu + 4);
      bf16* dg = &sBg[r * 80 + c]; bf16* du = &sBu[r * 80 + c];
      dg[0]=(bf16)a0.x; dg[1]=(bf16)a0.y; dg[2]=(bf16)a0.z; dg[3]=(bf16)a0.w;
      dg[4]=(bf16)a1.x; dg[5]=(bf16)a1.y; dg[6]=(bf16)a1.z; dg[7]=(bf16)a1.w;
      du[0]=(bf16)b0.x; du[1]=(bf16)b0.y; du[2]=(bf16)b0.z; du[3]=(bf16)b0.w;
      du[4]=(bf16)b1.x; du[5]=(bf16)b1.y; du[6]=(bf16)b1.z; du[7]=(bf16)b1.w;
      if (k0 + 32 < K) {
        __builtin_prefetch(Bg + (size_t)(k0 + 32 + r) * NI + (n0 + c), 0, 1);
        __builtin_prefetch(Bu + (size_t)(k0 + 32 + r) * NI + (n0 + c), 0, 1);
      }
    }
    async_wait0();
    __syncthreads();
    v16bf a0 = load_a_frag(sA, 40, mw,      lane);
    v16bf a1 = load_a_frag(sA, 40, mw + 16, lane);
    v16bf bg = load_b_frag(sBg, 80, nw, lane);
    v16bf bu = load_b_frag(sBu, 80, nw, lane);
    ag[0] = wmma_bf16(a0, bg, ag[0]);
    ag[1] = wmma_bf16(a1, bg, ag[1]);
    au[0] = wmma_bf16(a0, bu, au[0]);
    au[1] = wmma_bf16(a1, bu, au[1]);
    __syncthreads();
  }
  const int rb = (lane >> 4) * 8, cl = lane & 15;
#pragma unroll
  for (int mi = 0; mi < 2; ++mi)
#pragma unroll
    for (int r = 0; r < 8; ++r) {
      const int lr = mw + mi * 16 + rb + r;
      if (row0 + lr < count) {
        float v = siluf(ag[mi][r]) * au[mi][r];
        act[(size_t)(pairBase + row0 + lr) * NI + (n0 + nw + cl)] = (bf16)v;
      }
    }
}

// ---------------------------------------------------------------- expert down-proj + scatter-add
__launch_bounds__(256)
__global__ void k_moe_down(const bf16* __restrict__ act,
                           const int* __restrict__ idx, const float* __restrict__ pw,
                           const int* __restrict__ offs, const int* __restrict__ counts,
                           const float* __restrict__ Wd, long long wstride, int NI,
                           float* __restrict__ outAcc)
{
  const int e = blockIdx.z;
  const int pairBase = offs[e];
  const int count = counts[e];
  const int row0 = blockIdx.y * 64;
  if (row0 >= count) return;
  const int n0 = blockIdx.x * 64;
  const float* B = Wd + (size_t)e * wstride;
  __shared__ __align__(32) bf16 sA[64 * 40];
  __shared__ __align__(32) bf16 sB[32 * 80];
  __shared__ int sTok[64];
  __shared__ float sW[64];
  const int tid = threadIdx.x, lane = tid & 31, wid = tid >> 5;
  if (tid < 64) {
    int r = row0 + tid;
    int rc = (r < count) ? r : (count - 1);
    sTok[tid] = idx[pairBase + rc];
    sW[tid] = (r < count) ? pw[pairBase + rc] : 0.f;
  }
  __syncthreads();
  const int mw = (wid >> 2) * 32, nw = (wid & 3) * 16;
  v8f c0 = {}, c1 = {};
  for (int k0 = 0; k0 < NI; k0 += 32) {
    { // act tile 64x32 bf16: async DMA into LDS
      const int r = tid >> 2, kc = (tid & 3) * 8;
      const int rc = (row0 + r < count) ? (row0 + r) : (count - 1);
      const bf16* g = act + (size_t)(pairBase + rc) * NI + (k0 + kc);
      async_cp16(&sA[r * 40 + kc], g);
    }
    {
      const int r = tid >> 3, c = (tid & 7) * 8;
      const float* g = B + (size_t)(k0 + r) * H_ + (n0 + c);
      float4 f0 = *(const float4*)g, f1 = *(const float4*)(g + 4);
      bf16* d = &sB[r * 80 + c];
      d[0]=(bf16)f0.x; d[1]=(bf16)f0.y; d[2]=(bf16)f0.z; d[3]=(bf16)f0.w;
      d[4]=(bf16)f1.x; d[5]=(bf16)f1.y; d[6]=(bf16)f1.z; d[7]=(bf16)f1.w;
      if (k0 + 32 < NI)
        __builtin_prefetch(B + (size_t)(k0 + 32 + r) * H_ + (n0 + c), 0, 1);
    }
    async_wait0();
    __syncthreads();
    v16bf a0 = load_a_frag(sA, 40, mw,      lane);
    v16bf a1 = load_a_frag(sA, 40, mw + 16, lane);
    v16bf b  = load_b_frag(sB, 80, nw, lane);
    c0 = wmma_bf16(a0, b, c0);
    c1 = wmma_bf16(a1, b, c1);
    __syncthreads();
  }
  const int rb = (lane >> 4) * 8, cl = lane & 15;
#pragma unroll
  for (int mi = 0; mi < 2; ++mi) {
    v8f cc = mi ? c1 : c0;
#pragma unroll
    for (int r = 0; r < 8; ++r) {
      const int lr = mw + mi * 16 + rb + r;
      if (row0 + lr < count)
        atomicAdd(&outAcc[(size_t)sTok[lr] * H_ + (n0 + nw + cl)], cc[r] * sW[lr]);
    }
  }
}

// ---------------------------------------------------------------- rmsnorm (bf16 + optional f32 out)
__launch_bounds__(256)
__global__ void k_rmsnorm(const float* __restrict__ x, const float* __restrict__ w,
                          bf16* __restrict__ ob, float* __restrict__ of)
{
  const int row = blockIdx.x, tid = threadIdx.x;
  __shared__ float red[256];
  const float* xr = x + (size_t)row * H_;
  float v[4], s = 0.f;
#pragma unroll
  for (int i = 0; i < 4; ++i) { v[i] = xr[tid + 256 * i]; s += v[i] * v[i]; }
  red[tid] = s;
  __syncthreads();
  for (int st = 128; st > 0; st >>= 1) {
    if (tid < st) red[tid] += red[tid + st];
    __syncthreads();
  }
  const float inv = rsqrtf(red[0] * (1.f / H_) + 1e-5f);
#pragma unroll
  for (int i = 0; i < 4; ++i) {
    const int c = tid + 256 * i;
    float o = v[i] * w[c] * inv;
    ob[(size_t)row * H_ + c] = (bf16)o;
    if (of) of[(size_t)row * H_ + c] = o;
  }
}

// ---------------------------------------------------------------- mRoPE + qkv split
__launch_bounds__(256)
__global__ void k_rope_split(const float* __restrict__ qkv, const int* __restrict__ pos,
                             bf16* __restrict__ qh, bf16* __restrict__ kT, bf16* __restrict__ vh)
{
  const int s = blockIdx.x, tid = threadIdx.x;
  const float* row = qkv + (size_t)s * 1536;
#pragma unroll
  for (int it = 0; it < 3; ++it) {
    const int id = tid + it * 256;
    if (id < 512) {                       // q: 8 heads x 64 pairs
      const int h = id >> 6, d = id & 63;
      const float x1 = row[h * 128 + 2 * d], x2 = row[h * 128 + 2 * d + 1];
      const int axis = d < 22 ? 0 : (d < 44 ? 1 : 2);
      const float p = (float)pos[axis * S_ + s];
      const float fr = p * __powf(THETA_, -(2.f * d) / 128.f);
      float sn, cs; __sincosf(fr, &sn, &cs);
      bf16* q = qh + ((size_t)h * S_ + s) * D_;
      q[2 * d]     = (bf16)(x1 * cs - x2 * sn);
      q[2 * d + 1] = (bf16)(x1 * sn + x2 * cs);
    } else if (id < 640) {                // k: 2 kv heads x 64 pairs -> transposed [kv][D][S]
      const int kv = (id - 512) >> 6, d = (id - 512) & 63;
      const float x1 = row[1024 + kv * 128 + 2 * d], x2 = row[1024 + kv * 128 + 2 * d + 1];
      const int axis = d < 22 ? 0 : (d < 44 ? 1 : 2);
      const float p = (float)pos[axis * S_ + s];
      const float fr = p * __powf(THETA_, -(2.f * d) / 128.f);
      float sn, cs; __sincosf(fr, &sn, &cs);
      kT[((size_t)kv * D_ + 2 * d) * S_ + s]     = (bf16)(x1 * cs - x2 * sn);
      kT[((size_t)kv * D_ + 2 * d + 1) * S_ + s] = (bf16)(x1 * sn + x2 * cs);
    } else if (id < 768) {                // v: cast only
      const int kv = (id - 640) >> 6, d = (id - 640) & 63;
      bf16* vp = vh + ((size_t)kv * S_ + s) * D_;
      vp[2 * d]     = (bf16)row[1280 + kv * 128 + 2 * d];
      vp[2 * d + 1] = (bf16)row[1280 + kv * 128 + 2 * d + 1];
    }
  }
}

// ---------------------------------------------------------------- causal softmax row -> bf16 P
__launch_bounds__(256)
__global__ void k_softmax(const float* __restrict__ sc, bf16* __restrict__ P)
{
  const int i = blockIdx.x, tid = threadIdx.x;
  const float scale = 0.08838834764831845f;  // D^-0.5
  const float* rowp = sc + (size_t)i * S_;
  __shared__ float red[256];
  const int n = i + 1;
  float m = -1e30f;
  for (int j = tid; j < n; j += 256) m = fmaxf(m, rowp[j] * scale);
  red[tid] = m;
  __syncthreads();
  for (int st = 128; st > 0; st >>= 1) {
    if (tid < st) red[tid] = fmaxf(red[tid], red[tid + st]);
    __syncthreads();
  }
  m = red[0];
  __syncthreads();
  float ss = 0.f;
  for (int j = tid; j < n; j += 256) ss += __expf(rowp[j] * scale - m);
  red[tid] = ss;
  __syncthreads();
  for (int st = 128; st > 0; st >>= 1) {
    if (tid < st) red[tid] += red[tid + st];
    __syncthreads();
  }
  const float inv = 1.f / red[0];
  bf16* pr = P + (size_t)i * S_;
  for (int j = tid; j < S_; j += 256)
    pr[j] = (j < n) ? (bf16)(__expf(rowp[j] * scale - m) * inv) : (bf16)0.f;
}

// ---------------------------------------------------------------- router: sigmoid + biased top-6
__launch_bounds__(256)
__global__ void k_router(const float* __restrict__ X, const unsigned char* __restrict__ vmask,
                         const float* __restrict__ Wt, const float* __restrict__ Wv,
                         const float* __restrict__ bias,
                         int* __restrict__ topi, float* __restrict__ topw,
                         int* __restrict__ branch, int* __restrict__ counts)
{
  const int t = blockIdx.x, tid = threadIdx.x;
  __shared__ float sx[H_];
  __shared__ float part[256];
  __shared__ float sc[E_], sb[E_];
  for (int i = tid; i < H_; i += 256) sx[i] = X[(size_t)t * H_ + i];
  __syncthreads();
  const int b = vmask[t] ? 1 : 0;
  const float* W = b ? Wv : Wt;
  const int e = tid & 63, seg = tid >> 6;
  float p = 0.f;
  for (int i = 0; i < 256; ++i) {
    const int k = seg * 256 + i;
    p += sx[k] * W[(size_t)k * E_ + e];
  }
  part[tid] = p;
  __syncthreads();
  if (tid < 64) {
    float l = part[tid] + part[tid + 64] + part[tid + 128] + part[tid + 192];
    float s = 1.f / (1.f + __expf(-l));
    sc[tid] = s;
    sb[tid] = s + bias[b * E_ + tid];
  }
  __syncthreads();
  if (tid == 0) {
    int sel[TOPK_]; float w[TOPK_]; float sum = 0.f;
    for (int j = 0; j < TOPK_; ++j) {
      int bi = 0; float bv = -1e30f;
      for (int ee = 0; ee < E_; ++ee)
        if (sb[ee] > bv) { bv = sb[ee]; bi = ee; }
      sb[bi] = -1e30f;
      sel[j] = bi; w[j] = sc[bi]; sum += sc[bi];
    }
    const float inv = 1.f / sum;
    for (int j = 0; j < TOPK_; ++j) {
      topi[t * TOPK_ + j] = sel[j];
      topw[t * TOPK_ + j] = w[j] * inv;
      atomicAdd(&counts[b * E_ + sel[j]], 1);
    }
    branch[t] = b;
  }
}

__global__ void k_offsets(const int* __restrict__ counts, int* __restrict__ offs)
{
  if (threadIdx.x < 2) {
    const int b = threadIdx.x;
    int s = 0;
    for (int e = 0; e < E_; ++e) { offs[b * E_ + e] = s; s += counts[b * E_ + e]; }
  }
}

__global__ void k_fill(const int* __restrict__ topi, const float* __restrict__ topw,
                       const int* __restrict__ branch, const int* __restrict__ offs,
                       int* __restrict__ cnt2, int* __restrict__ pairTok, float* __restrict__ pairW)
{
  const int t = blockIdx.x * blockDim.x + threadIdx.x;
  if (t >= S_) return;
  const int b = branch[t];
  for (int j = 0; j < TOPK_; ++j) {
    const int e = topi[t * TOPK_ + j];
    const int pos = offs[b * E_ + e] + atomicAdd(&cnt2[b * E_ + e], 1);
    pairTok[b * CAP_ + pos] = t;
    pairW[b * CAP_ + pos] = topw[t * TOPK_ + j];
  }
}

// ---------------------------------------------------------------- elementwise
__global__ void k_add2(const float* __restrict__ a, const float* __restrict__ b, float* __restrict__ o)
{
  const size_t i = (size_t)blockIdx.x * blockDim.x + threadIdx.x;
  o[i] = a[i] + b[i];
}
__global__ void k_add3(const float* __restrict__ a, const float* __restrict__ b,
                       const float* __restrict__ c, float* __restrict__ o)
{
  const size_t i = (size_t)blockIdx.x * blockDim.x + threadIdx.x;
  o[i] = a[i] + b[i] + c[i];
}

// ---------------------------------------------------------------- host
extern "C" void kernel_launch(void* const* d_in, const int* in_sizes, int n_in,
                              void* d_out, int out_size, void* d_ws, size_t ws_size,
                              hipStream_t stream)
{
  (void)in_sizes; (void)n_in; (void)out_size; (void)ws_size;
  const float* hidden = (const float*)d_in[0];
  const int*   positions = (const int*)d_in[1];
  const unsigned char* vmask = (const unsigned char*)d_in[2];  // jnp bool -> 1 byte
  const float* w_qkv = (const float*)d_in[3];
  const float* w_o   = (const float*)d_in[4];
  const float* ln1   = (const float*)d_in[5];
  const float* ln2   = (const float*)d_in[6];
  const float* tgw   = (const float*)d_in[7];
  const float* vgw   = (const float*)d_in[8];
  const float* cbias = (const float*)d_in[9];
  const float* twg = (const float*)d_in[10];
  const float* twu = (const float*)d_in[11];
  const float* twd = (const float*)d_in[12];
  const float* vwg = (const float*)d_in[13];
  const float* vwu = (const float*)d_in[14];
  const float* vwd = (const float*)d_in[15];
  const float* swg = (const float*)d_in[16];
  const float* swu = (const float*)d_in[17];
  const float* swd = (const float*)d_in[18];
  float* out = (float*)d_out;

  char* ws = (char*)d_ws;
  size_t off = 0;
  auto alloc = [&](size_t bytes) -> void* {
    off = (off + 255) & ~(size_t)255;
    void* p = ws + off;
    off += bytes;
    return p;
  };
  float* hbuf    = (float*)alloc((size_t)S_ * H_ * 4);
  bf16*  xn1     = (bf16*) alloc((size_t)S_ * H_ * 2);
  float* qkv     = (float*)alloc((size_t)S_ * 1536 * 4);
  bf16*  qh      = (bf16*) alloc((size_t)NH_ * S_ * D_ * 2);
  bf16*  kT      = (bf16*) alloc((size_t)NKV_ * D_ * S_ * 2);
  bf16*  vh      = (bf16*) alloc((size_t)NKV_ * S_ * D_ * 2);
  float* scores  = (float*)alloc((size_t)S_ * S_ * 4);
  bf16*  P       = (bf16*) alloc((size_t)S_ * S_ * 2);
  bf16*  attnc   = (bf16*) alloc((size_t)S_ * H_ * 2);
  float* proj    = (float*)alloc((size_t)S_ * H_ * 4);
  bf16*  xn2     = (bf16*) alloc((size_t)S_ * H_ * 2);
  float* xn2f    = (float*)alloc((size_t)S_ * H_ * 4);
  bf16*  actS    = (bf16*) alloc((size_t)S_ * IS_ * 2);
  float* sharedO = (float*)alloc((size_t)S_ * H_ * 4);
  float* moeO    = (float*)alloc((size_t)S_ * H_ * 4);
  int*   topi    = (int*)  alloc((size_t)S_ * TOPK_ * 4);
  float* topw    = (float*)alloc((size_t)S_ * TOPK_ * 4);
  int*   branch  = (int*)  alloc((size_t)S_ * 4);
  int*   counts  = (int*)  alloc(2 * E_ * 4);
  int*   offs    = (int*)  alloc(2 * E_ * 4);
  int*   cnt2    = (int*)  alloc(2 * E_ * 4);
  int*   pairTok = (int*)  alloc((size_t)2 * CAP_ * 4);
  float* pairW   = (float*)alloc((size_t)2 * CAP_ * 4);
  bf16*  actT    = (bf16*) alloc((size_t)CAP_ * IT_ * 2);
  bf16*  actV    = (bf16*) alloc((size_t)CAP_ * IV_ * 2);

  hipMemsetAsync(counts, 0, 2 * E_ * 4, stream);
  hipMemsetAsync(cnt2, 0, 2 * E_ * 4, stream);
  hipMemsetAsync(moeO, 0, (size_t)S_ * H_ * 4, stream);

  // attention block
  k_rmsnorm<<<S_, 256, 0, stream>>>(hidden, ln1, xn1, nullptr);
  k_gemm<true, false, false><<<dim3(1536 / 64, S_ / 128), 256, 0, stream>>>(
      xn1, H_, w_qkv, 1536, qkv, 1536, S_, 1536, H_);
  k_rope_split<<<S_, 256, 0, stream>>>(qkv, positions, qh, kT, vh);
  for (int h = 0; h < NH_; ++h) {
    const int kv = h >> 2;  // NH/NKV = 4
    k_gemm<false, false, true><<<dim3(S_ / 64, S_ / 128), 256, 0, stream>>>(
        qh + (size_t)h * S_ * D_, D_, kT + (size_t)kv * D_ * S_, S_, scores, S_, S_, S_, D_);
    k_softmax<<<S_, 256, 0, stream>>>(scores, P);
    k_gemm<false, true, false><<<dim3(D_ / 64, S_ / 128), 256, 0, stream>>>(
        P, S_, vh + (size_t)kv * S_ * D_, D_, attnc + h * D_, H_, S_, D_, S_);
  }
  k_gemm<true, false, false><<<dim3(H_ / 64, S_ / 128), 256, 0, stream>>>(
      attnc, H_, w_o, H_, proj, H_, S_, H_, H_);
  k_add2<<<(S_ * H_) / 256, 256, 0, stream>>>(hidden, proj, hbuf);

  // MLP / MoE block
  k_rmsnorm<<<S_, 256, 0, stream>>>(hbuf, ln2, xn2, xn2f);
  k_router<<<S_, 256, 0, stream>>>(xn2f, vmask, tgw, vgw, cbias, topi, topw, branch, counts);
  k_offsets<<<1, 64, 0, stream>>>(counts, offs);
  k_fill<<<S_ / 256, 256, 0, stream>>>(topi, topw, branch, offs, cnt2, pairTok, pairW);

  // shared expert (dense)
  k_gateup<<<dim3(IS_ / 64, S_ / 64, 1), 256, 0, stream>>>(
      xn2, nullptr, nullptr, nullptr, S_, swg, swu, 0, actS, IS_, H_);
  k_gemm<true, false, false><<<dim3(H_ / 64, S_ / 128), 256, 0, stream>>>(
      actS, IS_, swd, H_, sharedO, H_, S_, H_, IS_);

  // routed experts: text branch
  k_gateup<<<dim3(IT_ / 64, 32, E_), 256, 0, stream>>>(
      xn2, pairTok, offs, counts, 0, twg, twu, (long long)H_ * IT_, actT, IT_, H_);
  k_moe_down<<<dim3(H_ / 64, 32, E_), 256, 0, stream>>>(
      actT, pairTok, pairW, offs, counts, twd, (long long)IT_ * H_, IT_, moeO);

  // routed experts: vision branch
  k_gateup<<<dim3(IV_ / 64, 32, E_), 256, 0, stream>>>(
      xn2, pairTok + CAP_, offs + E_, counts + E_, 0, vwg, vwu, (long long)H_ * IV_, actV, IV_, H_);
  k_moe_down<<<dim3(H_ / 64, 32, E_), 256, 0, stream>>>(
      actV, pairTok + CAP_, pairW + CAP_, offs + E_, counts + E_, vwd, (long long)IV_ * H_, IV_, moeO);

  // final residual sum
  k_add3<<<(S_ * H_) / 256, 256, 0, stream>>>(hbuf, sharedO, moeO, out);
}
